// CAttention_29884382446347
// MI455X (gfx1250) — compile-verified, matrix operand order checked by reading, not run on hardware
//
#include <hip/hip_runtime.h>
#include <hip/hip_bf16.h>

typedef __attribute__((ext_vector_type(16))) _Float16 v16h;
typedef __attribute__((ext_vector_type(8)))  _Float16 v8h;
typedef __attribute__((ext_vector_type(8)))  float    v8f;

#define L_DIM 4096   // H*W
#define C_DIM 128
#define EPSF  1e-7f

#define WMMA_F16(a, b, c) \
    __builtin_amdgcn_wmma_f32_16x16x32_f16(false, (a), false, (b), (short)0, (c), false, false)

// Load one 16x32 f16 A-fragment (or 32x16 B-fragment given B^T row-major).
// `row` must already point at (row_index)*ld + k0 + koff, 16B aligned.
// Lane layout (ISA 7.12.2): halves {0..7} at +0, {8..15} at +16 halves.
__device__ __forceinline__ v16h load_frag16(const _Float16* row) {
    v8h lo = *(const v8h*)(row);
    v8h hi = *(const v8h*)(row + 16);
    v16h r;
#pragma unroll
    for (int i = 0; i < 8; ++i) { r[i] = lo[i]; r[8 + i] = hi[i]; }
    return r;
}

// ---------------- prep: normalized k in two layouts -------------------------
// k[l,c] = (x[c,l]+eps)/||row||;  krm[l][c] (B^T for GEMM1), kT[c][l] (A for GEMM2)
__global__ void knorm_kernel(const float* __restrict__ x,
                             _Float16* __restrict__ krm,
                             _Float16* __restrict__ kT) {
    int l = blockIdx.x * blockDim.x + threadIdx.x;   // 4096 threads
    float ss = 0.0f;
#pragma unroll 4
    for (int c = 0; c < C_DIM; ++c) {
        float v = x[c * L_DIM + l] + EPSF;
        ss += v * v;
    }
    float inv = 1.0f / sqrtf(ss);
#pragma unroll 4
    for (int c = 0; c < C_DIM; ++c) {
        float v = (x[c * L_DIM + l] + EPSF) * inv;
        _Float16 h = (_Float16)v;
        krm[l * C_DIM + c] = h;
        kT[c * L_DIM + l]  = h;
    }
}

// ---------------- prep: 3x3 zero-padded box filter, transposed -------------
// yT[p][c] = sum_{3x3}(x[c, nbhd(p)])   (A matrix for GEMM1)
__global__ void box_kernel(const float* __restrict__ x,
                           _Float16* __restrict__ yT) {
    int idx = blockIdx.x * 256 + threadIdx.x;        // 128*4096 threads
    int c = idx >> 12;
    int p = idx & (L_DIM - 1);
    int i = p >> 6, j = p & 63;
    const float* xc = x + c * L_DIM;
    float s = 0.0f;
#pragma unroll
    for (int di = -1; di <= 1; ++di) {
        int ii = i + di;
        if (ii < 0 || ii > 63) continue;
#pragma unroll
        for (int dj = -1; dj <= 1; ++dj) {
            int jj = j + dj;
            if (jj < 0 || jj > 63) continue;
            s += xc[ii * 64 + jj];
        }
    }
    yT[p * C_DIM + c] = (_Float16)s;
}

// ---------------- GEMM1: S[p][l] = sum_c yT[p][c]*krm[l][c] ----------------
// block = 8 waves; each wave one 16(M) x 32(N) strip (2 tiles, A reused);
// grid = (256 p-tiles, 16 l-groups)
__global__ void gemm1_kernel(const _Float16* __restrict__ yT,
                             const _Float16* __restrict__ krm,
                             float* __restrict__ S) {
    int wave = threadIdx.x >> 5;
    int lane = threadIdx.x & 31;
    int m0 = blockIdx.x * 16;                    // p tile
    int n0 = (blockIdx.y * 8 + wave) * 32;       // 2 l tiles
    int r    = lane & 15;
    int koff = (lane >> 4) << 3;                 // 0 or 8

    const _Float16* arow  = yT  + (m0 + r) * C_DIM + koff;
    const _Float16* brow0 = krm + (n0 + r) * C_DIM + koff;
    const _Float16* brow1 = krm + (n0 + 16 + r) * C_DIM + koff;

    v8f acc0 = {}, acc1 = {};
#pragma unroll
    for (int kt = 0; kt < C_DIM / 32; ++kt) {    // K = 128 -> 4 steps
        v16h a  = load_frag16(arow  + kt * 32);
        v16h b0 = load_frag16(brow0 + kt * 32);
        v16h b1 = load_frag16(brow1 + kt * 32);
        acc0 = WMMA_F16(a, b0, acc0);
        acc1 = WMMA_F16(a, b1, acc1);
    }
    int n  = lane & 15;
    int mb = (lane >> 4) << 3;
    float* dst = S + (size_t)(m0 + mb) * L_DIM + n0 + n;
#pragma unroll
    for (int i = 0; i < 8; ++i) {
        dst[(size_t)i * L_DIM]      = acc0[i];
        dst[(size_t)i * L_DIM + 16] = acc1[i];
    }
}

// ---------------- softmax over l per row p; write f16 att in place ----------
// att[p][l] (f16, ld 8192 halves) overwrites the start of S's f32 row; the
// f32 row is cached in LDS first, so the overwrite is safe.
__global__ void softmax_kernel(float* __restrict__ S) {
    __shared__ float row[L_DIM];
    __shared__ float red[256];
    int p = blockIdx.x;
    float* src = S + (size_t)p * L_DIM;

    float mx = -3.0e38f;
    for (int i = threadIdx.x; i < L_DIM; i += 256) {
        float v = src[i];
        row[i] = v;
        mx = fmaxf(mx, v);
    }
    red[threadIdx.x] = mx;
    __syncthreads();
    for (int s = 128; s > 0; s >>= 1) {
        if (threadIdx.x < s) red[threadIdx.x] = fmaxf(red[threadIdx.x], red[threadIdx.x + s]);
        __syncthreads();
    }
    mx = red[0];
    __syncthreads();

    float sum = 0.0f;
    for (int i = threadIdx.x; i < L_DIM; i += 256) sum += expf(row[i] - mx);
    red[threadIdx.x] = sum;
    __syncthreads();
    for (int s = 128; s > 0; s >>= 1) {
        if (threadIdx.x < s) red[threadIdx.x] += red[threadIdx.x + s];
        __syncthreads();
    }
    float inv = 1.0f / red[0];

    _Float16* dst = (_Float16*)src;              // in-place f16 row
    for (int i = threadIdx.x; i < L_DIM; i += 256)
        dst[i] = (_Float16)(expf(row[i] - mx) * inv);
}

// ---------------- GEMM2 + epilogue ------------------------------------------
// R[c][p] = sum_l kT[c][l]*att[p][l];  out = R*(1-m)/9 + x*m
// block = 8 waves; wave w owns c-tile w and a 32-wide p strip (2 N tiles,
// A fragment reused); K = 4096 -> 128 WMMA steps, unrolled x2, with
// global_prefetch on the three fragment streams.
__global__ void gemm2_kernel(const _Float16* __restrict__ kT,
                             const _Float16* __restrict__ att,  // ld 8192 halves
                             const float* __restrict__ x,
                             const float* __restrict__ mask,
                             float* __restrict__ out) {
    int wave = threadIdx.x >> 5;
    int lane = threadIdx.x & 31;
    int m0 = wave * 16;                          // c tile
    int n0 = blockIdx.x * 32;                    // 2 p tiles
    int r    = lane & 15;
    int koff = (lane >> 4) << 3;

    const _Float16* arow  = kT  + (size_t)(m0 + r) * L_DIM + koff;
    const _Float16* brow0 = att + (size_t)(n0 + r) * 8192  + koff;
    const _Float16* brow1 = att + (size_t)(n0 + 16 + r) * 8192 + koff;

    v8f acc0 = {}, acc1 = {};
    for (int kt = 0; kt < L_DIM / 32; kt += 2) { // 64 iterations, 4 wmma each
        if (kt + 8 < L_DIM / 32) {               // L2 -> WGP prefetch, ~4 iters ahead
            __builtin_prefetch(arow  + (kt + 8) * 32, 0, 0);
            __builtin_prefetch(brow0 + (kt + 8) * 32, 0, 0);
            __builtin_prefetch(brow1 + (kt + 8) * 32, 0, 0);
        }
#pragma unroll
        for (int u = 0; u < 2; ++u) {
            v16h a  = load_frag16(arow  + (kt + u) * 32);
            v16h b0 = load_frag16(brow0 + (kt + u) * 32);
            v16h b1 = load_frag16(brow1 + (kt + u) * 32);
            acc0 = WMMA_F16(a, b0, acc0);
            acc1 = WMMA_F16(a, b1, acc1);
        }
    }
    int n  = lane & 15;
    int mb = (lane >> 4) << 3;
    int p0 = n0 + n;
    int p1 = n0 + 16 + n;
    float mv0 = mask[p0], mv1 = mask[p1];
    float sc0 = (1.0f - mv0) * (1.0f / 9.0f);
    float sc1 = (1.0f - mv1) * (1.0f / 9.0f);
#pragma unroll
    for (int i = 0; i < 8; ++i) {
        int c = m0 + mb + i;
        out[(size_t)c * L_DIM + p0] = acc0[i] * sc0 + x[(size_t)c * L_DIM + p0] * mv0;
        out[(size_t)c * L_DIM + p1] = acc1[i] * sc1 + x[(size_t)c * L_DIM + p1] * mv1;
    }
}

// ---------------------------------------------------------------------------
extern "C" void kernel_launch(void* const* d_in, const int* in_sizes, int n_in,
                              void* d_out, int out_size, void* d_ws, size_t ws_size,
                              hipStream_t stream) {
    const float* fg   = (const float*)d_in[0];   // (4,128,64,64)
    const float* mask = (const float*)d_in[1];   // (4,1,64,64)
    float* outp = (float*)d_out;                 // (4,128,64,64)

    // Workspace layout (reused across the 4 sequential batches):
    //   S   : 4096*4096 f32  = 64 MB   (softmax rewrites each row head as f16 att)
    //   yT  : 4096*128  f16  =  1 MB
    //   krm : 4096*128  f16  =  1 MB
    //   kT  : 128*4096  f16  =  1 MB
    char* ws = (char*)d_ws;
    float*    S   = (float*)ws;
    _Float16* yT  = (_Float16*)(ws + (size_t)L_DIM * L_DIM * 4);
    _Float16* krm = yT  + (size_t)L_DIM * C_DIM;
    _Float16* kT  = krm + (size_t)L_DIM * C_DIM;

    for (int b = 0; b < 4; ++b) {
        const float* x  = fg   + (size_t)b * C_DIM * L_DIM;
        const float* mk = mask + (size_t)b * L_DIM;
        float*       o  = outp + (size_t)b * C_DIM * L_DIM;

        knorm_kernel<<<L_DIM / 256, 256, 0, stream>>>(x, krm, kT);
        box_kernel<<<(C_DIM * L_DIM) / 256, 256, 0, stream>>>(x, yT);
        gemm1_kernel<<<dim3(L_DIM / 16, L_DIM / 256), 256, 0, stream>>>(yT, krm, S);
        softmax_kernel<<<L_DIM, 256, 0, stream>>>(S);
        gemm2_kernel<<<L_DIM / 32, 256, 0, stream>>>(kT, (const _Float16*)S, x, mk, o);
    }
}
